// CascadeDEDBackbone_75746043232708
// MI455X (gfx1250) — matrix operand order repaired and checked by reading, not compiled, not run on hardware
//
#include <hip/hip_runtime.h>
#include <hip/hip_bf16.h>

typedef __bf16 bf16;
typedef __attribute__((ext_vector_type(16))) __bf16 v16bf;
typedef __attribute__((ext_vector_type(8)))  __bf16 v8bf;
typedef __attribute__((ext_vector_type(8)))  float   v8f;

#define BN_SCALE 0.99950037f   // 1/sqrt(1 + 1e-3)
#define CCH 128                // channel count, fixed for the whole network

// ---------------------------------------------------------------------------
// Weight packers: OIHW f32 -> [tap][ci][co] bf16   (tap = ky*KW + kx)
// ---------------------------------------------------------------------------
__global__ __launch_bounds__(256) void pack_oihw(const float* __restrict__ w,
                                                 bf16* __restrict__ wp,
                                                 int KHW) {
  int i = blockIdx.x * 256 + threadIdx.x;
  int total = CCH * CCH * KHW;
  if (i >= total) return;
  int co = i / (CCH * KHW);
  int r  = i - co * (CCH * KHW);
  int ci = r / KHW;
  int t  = r - ci * KHW;
  wp[((long)t * CCH + ci) * CCH + co] = (bf16)w[i];
}

// ConvTranspose weights: [Cin][Cout][2][2] f32 -> [tap][ci][co] bf16
__global__ __launch_bounds__(256) void pack_iohw22(const float* __restrict__ w,
                                                   bf16* __restrict__ wp) {
  int i = blockIdx.x * 256 + threadIdx.x;
  int total = CCH * CCH * 4;
  if (i >= total) return;
  int ci = i >> 9;            // / (CCH*4)
  int r  = i & 511;
  int co = r >> 2;
  int t  = r & 3;
  wp[((long)t * CCH + ci) * CCH + co] = (bf16)w[i];
}

__global__ __launch_bounds__(256) void cvt_f32_bf16(const float* __restrict__ x,
                                                    bf16* __restrict__ y, int n) {
  int i = blockIdx.x * 256 + threadIdx.x;
  if (i < n) y[i] = (bf16)x[i];
}

// ---------------------------------------------------------------------------
// Implicit-GEMM conv using V_WMMA_F32_16X16X32_BF16.
//   C[m][n] = sum_k A[m][k] * B[k][n],  m = (b,oy,ox) pixel, n = out channel,
//   k = tap*128 + ci.  K stepped 32 at a time; c0 loop fully unrolled (x4).
// Block: 256 threads = 8 waves, tile 128(M) x 128(N); wave = 64(M) x 32(N).
// All spatial dims are powers of two -> shift/mask addressing only.
// MODE: 0 relu(a*s)->bf16 | 1 relu(a*s+Res)->bf16 | 2 a*s->bf16
//       3 (relu(a*s)+Res)*s->bf16 | 4 (relu(a*s)+Res)*s->f32
// DECONV: scatter output/res to (2y+dk, 2x+dl) of a (2H,2W) tensor.
// ---------------------------------------------------------------------------
template <int MODE, bool DECONV>
__global__ __launch_bounds__(256) void conv_wmma(
    const bf16* __restrict__ X, const bf16* __restrict__ Wp,
    const bf16* __restrict__ Res, void* __restrict__ Out,
    int Hin, int Win, int in_hw_shift,     // log2(Hin*Win)
    int hw_shift, int wout_shift,          // log2(Hout*Wout), log2(Wout)
    int KH, int KW, int pad, int stride, int dk, int dl) {

  __shared__ bf16 As[128][40];   // [m][k], padded row stride (80B, 16B aligned)
  __shared__ bf16 Bs[128][40];   // [n][k], transposed weights

  const int tid  = threadIdx.x;
  const int lane = tid & 31;
  const int wave = tid >> 5;
  const int wm   = wave & 1;   // 2 waves in M
  const int wn   = wave >> 1;  // 4 waves in N
  const int m0   = blockIdx.x * 128;
  const int HWm1 = (1 << hw_shift) - 1;
  const int Wm1  = (1 << wout_shift) - 1;

  v8f acc[4][2];
#pragma unroll
  for (int i = 0; i < 4; ++i)
#pragma unroll
    for (int j = 0; j < 2; ++j)
#pragma unroll
      for (int r = 0; r < 8; ++r) acc[i][j][r] = 0.0f;

  const int ntap = KH * KW;
  int ky = 0, kx = 0;
  for (int tap = 0; tap < ntap; ++tap) {
#pragma unroll
    for (int c0 = 0; c0 < 128; c0 += 32) {
      // ---- stage A tile: 128 pixels x 32 input channels (im2col gather) ----
#pragma unroll
      for (int w = 0; w < 2; ++w) {
        int wi  = tid + 256 * w;
        int cil = wi >> 4;        // 0..31 local ci
        int seg = wi & 15;        // 16 segments of 8 contiguous pixels
        int mb  = m0 + seg * 8;
        int b   = mb >> hw_shift;
        int rem = mb & HWm1;
        int oy  = rem >> wout_shift;
        int oxb = rem & Wm1;
        int iy  = oy * stride + ky - pad;
        bool rowok = (iy >= 0) && (iy < Hin);
        long basec = ((long)(b * CCH + c0 + cil) << in_hw_shift) + (long)iy * Win;
#pragma unroll
        for (int e = 0; e < 8; ++e) {
          int ix  = (oxb + e) * stride + kx - pad;
          bool ok = rowok && (ix >= 0) && (ix < Win);
          long g  = ok ? (basec + ix) : 0;
          bf16 v  = X[g];
          As[seg * 8 + e][cil] = ok ? v : (bf16)0.0f;
        }
      }
      // ---- stage B tile: 32 k-rows x 128 out channels, store [n][k] ----
#pragma unroll
      for (int w = 0; w < 2; ++w) {
        int wi  = tid + 256 * w;
        int kk  = wi >> 4;          // 0..31
        int coc = (wi & 15) * 8;    // 8 contiguous out channels
        const bf16* src = Wp + ((long)(tap * CCH + c0 + kk)) * CCH + coc;
        v8bf vv = *(const v8bf*)src;
#pragma unroll
        for (int j = 0; j < 8; ++j) Bs[coc + j][kk] = vv[j];
        // prefetch next tap's weight line into L2 (global_prefetch_b8)
        if (tap + 1 < ntap)
          __builtin_prefetch(src + (long)CCH * CCH, 0, 1);
      }
      __syncthreads();

      // ---- WMMA: 8 tiles of 16x16, K=32 ----
      const int lh  = lane & 15;
      const int kbA = (lane < 16) ? 0 : 8;   // A halves 0-7 -> K kbA.., 8-15 -> 16+kbA..
      const int kbB = (lane < 16) ? 0 : 8;   // B halves 0-7 -> K kbB.., 8-15 -> 16+kbB..

      v16bf bfrag[2];
#pragma unroll
      for (int nt = 0; nt < 2; ++nt) {
        const bf16* bp = &Bs[wn * 32 + nt * 16 + lh][kbB];
        v8bf lo = *(const v8bf*)bp;
        v8bf hi = *(const v8bf*)(bp + 16);
#pragma unroll
        for (int j = 0; j < 8; ++j) { bfrag[nt][j] = lo[j]; bfrag[nt][j + 8] = hi[j]; }
      }
#pragma unroll
      for (int mt = 0; mt < 4; ++mt) {
        const bf16* ap = &As[wm * 64 + mt * 16 + lh][kbA];
        v8bf lo = *(const v8bf*)ap;
        v8bf hi = *(const v8bf*)(ap + 16);
        v16bf afrag;
#pragma unroll
        for (int j = 0; j < 8; ++j) { afrag[j] = lo[j]; afrag[j + 8] = hi[j]; }
#pragma unroll
        for (int nt = 0; nt < 2; ++nt) {
          acc[mt][nt] = __builtin_amdgcn_wmma_f32_16x16x32_bf16(
              false, afrag, false, bfrag[nt], (short)0, acc[mt][nt], false, false);
        }
      }
      __syncthreads();
    }
    // advance (ky,kx) without division
    if (++kx == KW) { kx = 0; ++ky; }
  }

  // ---- fused epilogue (shift/mask addressing; template-specialized) ----
  const float s = BN_SCALE;
  bf16*  ob = (bf16*)Out;
  float* of = (float*)Out;
  const int mhalf = (lane < 16) ? 0 : 8;   // C/D layout: M = r + (lane>=16 ? 8 : 0)
#pragma unroll
  for (int mt = 0; mt < 4; ++mt) {
#pragma unroll
    for (int nt = 0; nt < 2; ++nt) {
      int n = wn * 32 + nt * 16 + (lane & 15);
#pragma unroll
      for (int r = 0; r < 8; ++r) {
        int m   = m0 + wm * 64 + mt * 16 + mhalf + r;
        int b   = m >> hw_shift;
        int rem = m & HWm1;
        long oidx;
        if (DECONV) {
          int oy = rem >> wout_shift;
          int ox = rem & Wm1;
          oidx = ((long)(b * CCH + n) << (hw_shift + 2)) +
                 ((long)(2 * oy + dk) << (wout_shift + 1)) + 2 * ox + dl;
        } else {
          oidx = ((long)(b * CCH + n) << hw_shift) + rem;
        }
        float v = acc[mt][nt][r] * s;
        if (MODE == 0) {
          v = fmaxf(v, 0.0f);
          ob[oidx] = (bf16)v;
        } else if (MODE == 1) {
          v += (float)Res[oidx];
          v = fmaxf(v, 0.0f);
          ob[oidx] = (bf16)v;
        } else if (MODE == 2) {
          ob[oidx] = (bf16)v;
        } else if (MODE == 3) {
          v = fmaxf(v, 0.0f);
          v = (v + (float)Res[oidx]) * s;
          ob[oidx] = (bf16)v;
        } else {
          v = fmaxf(v, 0.0f);
          v = (v + (float)Res[oidx]) * s;
          of[oidx] = v;
        }
      }
    }
  }
}

// ---------------------------------------------------------------------------
// Host orchestration
// ---------------------------------------------------------------------------
extern "C" void kernel_launch(void* const* d_in, const int* in_sizes, int n_in,
                              void* d_out, int out_size, void* d_ws, size_t ws_size,
                              hipStream_t stream) {
  (void)in_sizes; (void)n_in; (void)out_size; (void)ws_size;
  const int B = 2, C = CCH;

  char* ws = (char*)d_ws;
  size_t off = 0;
  auto take = [&](size_t bytes) -> void* {
    void* p = ws + off;
    off = (off + bytes + 255) & ~(size_t)255;
    return p;
  };

  const size_t FULL = (size_t)B * C * 256 * 256 * sizeof(bf16);  // 32 MB
  const size_t HALF = (size_t)B * C * 128 * 128 * sizeof(bf16);  //  8 MB
  bf16* P  = (bf16*)take(FULL);
  bf16* Q  = (bf16*)take(FULL);
  bf16* R  = (bf16*)take(FULL);
  bf16* F0 = (bf16*)take(FULL);
  bf16* F1 = (bf16*)take(HALF);
  bf16* ID = (bf16*)take(HALF);

  bf16* w33[12];
  const int idx33[12] = {1, 2, 3, 4, 5, 6, 8, 9, 10, 11, 13, 14};
  for (int i = 0; i < 12; ++i) w33[i] = (bf16*)take((size_t)9 * C * C * sizeof(bf16));
  bf16* wds0  = (bf16*)take((size_t)C * C * sizeof(bf16));
  bf16* wds1  = (bf16*)take((size_t)C * C * sizeof(bf16));
  bf16* wdec0 = (bf16*)take((size_t)4 * C * C * sizeof(bf16));
  bf16* wdec1 = (bf16*)take((size_t)4 * C * C * sizeof(bf16));

  // pack weights
  for (int i = 0; i < 12; ++i)
    pack_oihw<<<(C * C * 9 + 255) / 256, 256, 0, stream>>>(
        (const float*)d_in[idx33[i]], w33[i], 9);
  pack_oihw<<<(C * C + 255) / 256, 256, 0, stream>>>((const float*)d_in[7],  wds0, 1);
  pack_oihw<<<(C * C + 255) / 256, 256, 0, stream>>>((const float*)d_in[12], wds1, 1);
  pack_iohw22<<<(C * C * 4 + 255) / 256, 256, 0, stream>>>((const float*)d_in[15], wdec0);
  pack_iohw22<<<(C * C * 4 + 255) / 256, 256, 0, stream>>>((const float*)d_in[16], wdec1);

  // convert input activations to bf16
  const int NX = B * C * 256 * 256;
  cvt_f32_bf16<<<(NX + 255) / 256, 256, 0, stream>>>((const float*)d_in[0], P, NX);

  auto ilog2 = [](int v) { int s = 0; while ((1 << s) < v) ++s; return s; };

  auto conv = [&](const bf16* X, const bf16* W, const bf16* Rs, void* O,
                  int Hin, int Win, int Ho, int Wo, int KH, int KW,
                  int pad, int stride, int mode, int dkl) {
    int M = B * Ho * Wo;                       // always a multiple of 128 here
    dim3 grid(M / 128), blk(256);
    int ihs = ilog2(Hin * Win);
    int hs  = ilog2(Ho * Wo);
    int wsft = ilog2(Wo);
    int dk = (dkl >= 0) ? (dkl >> 1) : 0;
    int dl = (dkl >= 0) ? (dkl & 1) : 0;
    switch (mode) {
      case 0: conv_wmma<0, false><<<grid, blk, 0, stream>>>(
                  X, W, Rs, O, Hin, Win, ihs, hs, wsft, KH, KW, pad, stride, dk, dl); break;
      case 1: conv_wmma<1, false><<<grid, blk, 0, stream>>>(
                  X, W, Rs, O, Hin, Win, ihs, hs, wsft, KH, KW, pad, stride, dk, dl); break;
      case 2: conv_wmma<2, false><<<grid, blk, 0, stream>>>(
                  X, W, Rs, O, Hin, Win, ihs, hs, wsft, KH, KW, pad, stride, dk, dl); break;
      case 3: conv_wmma<3, true><<<grid, blk, 0, stream>>>(
                  X, W, Rs, O, Hin, Win, ihs, hs, wsft, KH, KW, pad, stride, dk, dl); break;
      default: conv_wmma<4, true><<<grid, blk, 0, stream>>>(
                  X, W, Rs, O, Hin, Win, ihs, hs, wsft, KH, KW, pad, stride, dk, dl); break;
    }
  };

  // encoder level 0 (256x256)
  conv(P, w33[0], nullptr, Q, 256, 256, 256, 256, 3, 3, 1, 1, 0, -1);
  conv(Q, w33[1], P,       R, 256, 256, 256, 256, 3, 3, 1, 1, 1, -1);
  conv(R, w33[2], nullptr, Q, 256, 256, 256, 256, 3, 3, 1, 1, 0, -1);
  conv(Q, w33[3], R,      F0, 256, 256, 256, 256, 3, 3, 1, 1, 1, -1);  // f0 skip

  // encoder level 1 (stride-2 down to 128x128)
  conv(F0, w33[4], nullptr, Q, 256, 256, 128, 128, 3, 3, 1, 2, 0, -1);
  conv(F0, wds0,  nullptr, ID, 256, 256, 128, 128, 1, 1, 0, 2, 2, -1);
  conv(Q,  w33[5], ID,      R, 128, 128, 128, 128, 3, 3, 1, 1, 1, -1);
  conv(R,  w33[6], nullptr, Q, 128, 128, 128, 128, 3, 3, 1, 1, 0, -1);
  conv(Q,  w33[7], R,      F1, 128, 128, 128, 128, 3, 3, 1, 1, 1, -1); // f1 skip

  // encoder level 2 (stride-2 down to 64x64)
  conv(F1, w33[8], nullptr, Q, 128, 128, 64, 64, 3, 3, 1, 2, 0, -1);
  conv(F1, wds1,  nullptr, ID, 128, 128, 64, 64, 1, 1, 0, 2, 2, -1);
  conv(Q,  w33[9], ID,      R, 64, 64, 64, 64, 3, 3, 1, 1, 1, -1);
  conv(R,  w33[10], nullptr, Q, 64, 64, 64, 64, 3, 3, 1, 1, 0, -1);
  conv(Q,  w33[11], R,       P, 64, 64, 64, 64, 3, 3, 1, 1, 1, -1);

  // decoder stage 0: (relu(deconv(P)*s) + f1)*s -> Q (128x128, bf16)
  for (int t = 0; t < 4; ++t)
    conv(P, wdec0 + (size_t)t * C * C, F1, Q, 64, 64, 64, 64, 1, 1, 0, 1, 3, t);
  // decoder stage 1: (relu(deconv(Q)*s) + f0)*s -> d_out (256x256, f32)
  for (int t = 0; t < 4; ++t)
    conv(Q, wdec1 + (size_t)t * C * C, F0, d_out, 128, 128, 128, 128, 1, 1, 0, 1, 4, t);
}
